// FFMCell_20607253086803
// MI455X (gfx1250) — compile-verified, matrix operand order checked by reading, not run on hardware
//
#include <hip/hip_runtime.h>

#define T_DIM 32768
#define DIN   128
#define MDIM  64
#define CDIM  16
#define ODIM  128
#define NCAT  384            // [pre(64) | gin(64) | gout(128) | skip(128)]
#define KMIX  2048           // 2*M*C
#define CL    128            // scan chunk length
#define CH    (T_DIM / CL)   // 256 chunks
#define NPAIR (MDIM * CDIM)  // 1024 (m,c) pairs

typedef __attribute__((__ext_vector_type__(16))) __bf16 v16bf;
typedef __attribute__((__ext_vector_type__(8)))  float  v8f;

union BF16Vec { v16bf v; uint4 q[2]; };

static __device__ __forceinline__ float sigmoidf_(float x) {
  return 1.0f / (1.0f + __expf(-x));
}

// ---------------------------------------------------------------------------
// CDNA5 async global->LDS path (ASYNCcnt). Guarded: falls back to plain
// VGPR-bounce copies if this toolchain doesn't declare the builtins.
// Builtin signature (from the round-2 diagnostic): first param is
// int4-vector pointer in the global address space; LDS param is AS3.
// ---------------------------------------------------------------------------
#if defined(__HIP_DEVICE_COMPILE__) && \
    __has_builtin(__builtin_amdgcn_global_load_async_to_lds_b128) && \
    __has_builtin(__builtin_amdgcn_s_wait_asynccnt)
#define USE_ASYNC_LDS 1
#else
#define USE_ASYNC_LDS 0
#endif

#if USE_ASYNC_LDS
typedef int v4i_ __attribute__((__vector_size__(16)));
typedef v4i_ __attribute__((address_space(1))) as1_v4i;
typedef v4i_ __attribute__((address_space(3))) as3_v4i;
static __device__ __forceinline__ void async_ld16(const uint4* g, uint4* l) {
  // generic->AS1: same value; generic LDS ptr low 32 bits == LDS offset (AS3)
  as1_v4i* gp = (as1_v4i*)(unsigned long long)g;
  as3_v4i* lp = (as3_v4i*)(unsigned int)(unsigned long long)l;
  __builtin_amdgcn_global_load_async_to_lds_b128(gp, lp, 0, 0);
}
#endif

// ---------------------------------------------------------------------------
// Weight swizzle: pack W into the exact WMMA B-operand (32x16 bf16) lane
// layout so GEMM inner loops read one contiguous 32B v16bf per lane from LDS.
// value B[k][n] within a (ntile,kstep):  kk=k%32, nn=n%16
//   lane = 16*((kk>>3)&1) + nn ; elem = (kk&7) + 8*(kk>>4)
// ---------------------------------------------------------------------------
__global__ __launch_bounds__(256) void k_swizzle_cat(
    const float* __restrict__ Wpre, const float* __restrict__ Wgin,
    const float* __restrict__ Wgout, const float* __restrict__ Wskip,
    __bf16* __restrict__ dst /* [24 nt][4 ks][32][16] */)
{
  const int idx = blockIdx.x * 256 + threadIdx.x;   // over 128*384
  const int k = idx / NCAT, col = idx % NCAT;
  float v;
  if (col < 64)        v = Wpre [k * 64  + col];
  else if (col < 128)  v = Wgin [k * 64  + (col - 64)];
  else if (col < 256)  v = Wgout[k * 128 + (col - 128)];
  else                 v = Wskip[k * 128 + (col - 256)];
  const int nt = col >> 4, ks = k >> 5, kk = k & 31, nn = col & 15;
  const int lane = (((kk >> 3) & 1) << 4) + nn;
  const int elem = (kk & 7) + ((kk >> 4) << 3);
  dst[(((nt * 4 + ks) * 32 + lane) << 4) + elem] = (__bf16)v;
}

__global__ __launch_bounds__(256) void k_swizzle_mix(
    const float* __restrict__ Wmix, __bf16* __restrict__ dst /* [64 ks][8 nt][32][16] */)
{
  const int idx = blockIdx.x * 256 + threadIdx.x;   // over 2048*128
  const int k = idx >> 7, col = idx & 127;
  const float v = Wmix[(size_t)k * ODIM + col];
  const int nt = col >> 4, ks = k >> 5, kk = k & 31, nn = col & 15;
  const int lane = (((kk >> 3) & 1) << 4) + nn;
  const int elem = (kk & 7) + ((kk >> 4) << 3);
  dst[(((ks * 8 + nt) * 32 + lane) << 4) + elem] = (__bf16)v;
}

// ---------------------------------------------------------------------------
// GEMM1: y[T,384] = x[T,128] @ Wcat[128,384] + bias. 8 waves/block, wave owns
// 16 rows. A (16x128) in registers; 4 weight slabs of 6 ntiles, double-
// buffered LDS (2 x 24 KB) fed by async global->LDS when available.
// ---------------------------------------------------------------------------
__global__ __launch_bounds__(256) void k_gemm1(
    const float* __restrict__ x, const __bf16* __restrict__ Wsw,
    const float* __restrict__ bpre, const float* __restrict__ bgin,
    const float* __restrict__ bgout, const float* __restrict__ bskip,
    float* __restrict__ y)
{
  __shared__ __bf16 lds[2][6 * 4 * 32 * 16];   // 2 x 24 KB
  const int tid = threadIdx.x, wave = tid >> 5, lane = tid & 31;
  const int rtile = lane & 15, hi = lane >> 4;
  const int rowA = blockIdx.x * 128 + wave * 16 + rtile;

#if USE_ASYNC_LDS
  {
    const uint4* src = (const uint4*)Wsw;
    uint4* dst = (uint4*)&lds[0][0];
    #pragma unroll
    for (int i = 0; i < 6; ++i) async_ld16(src + tid + 256 * i, dst + tid + 256 * i);
  }
#endif

  BF16Vec A[4];
  const float* xr = x + (size_t)rowA * DIN;
  #pragma unroll
  for (int ks = 0; ks < 4; ++ks) {
    const int k0 = ks * 32 + hi * 8;
    #pragma unroll
    for (int i = 0; i < 8; ++i) {
      A[ks].v[i]     = (__bf16)xr[k0 + i];
      A[ks].v[8 + i] = (__bf16)xr[k0 + 16 + i];
    }
  }

  const int outRow0 = blockIdx.x * 128 + wave * 16 + hi * 8;

  for (int slab = 0; slab < 4; ++slab) {
    const int cur = slab & 1;
#if USE_ASYNC_LDS
    __builtin_amdgcn_s_wait_asynccnt(0);   // our slab's data has landed
    __syncthreads();                       // everyone's data visible
    if (slab + 1 < 4) {                    // prefetch next slab, overlap WMMAs
      const uint4* src = (const uint4*)(Wsw + (size_t)(slab + 1) * (6 * 4 * 32 * 16));
      uint4* dst = (uint4*)&lds[cur ^ 1][0];
      #pragma unroll
      for (int i = 0; i < 6; ++i) async_ld16(src + tid + 256 * i, dst + tid + 256 * i);
    }
#else
    __syncthreads();
    {
      const uint4* src = (const uint4*)(Wsw + (size_t)slab * (6 * 4 * 32 * 16));
      uint4* dst = (uint4*)&lds[cur][0];
      #pragma unroll
      for (int i = 0; i < 6; ++i) dst[tid + 256 * i] = src[tid + 256 * i];
    }
    __syncthreads();
#endif

    for (int nt = 0; nt < 6; ++nt) {
      v8f acc = {};
      #pragma unroll
      for (int ks = 0; ks < 4; ++ks) {
        const v16bf B = *(const v16bf*)&lds[cur][((nt * 4 + ks) * 32 + lane) << 4];
        acc = __builtin_amdgcn_wmma_f32_16x16x32_bf16(false, A[ks].v, false, B,
                                                      (short)0, acc, false, false);
      }
      const int col = (slab * 6 + nt) * 16 + rtile;
      float bias;
      if (col < 64)        bias = bpre[col];
      else if (col < 128)  bias = bgin[col - 64];
      else if (col < 256)  bias = bgout[col - 128];
      else                 bias = bskip[col - 256];
      #pragma unroll
      for (int i = 0; i < 8; ++i)
        y[(size_t)(outRow0 + i) * NCAT + col] = acc[i] + bias;
    }
  }
}

// ---------------------------------------------------------------------------
// Scan pass A: per-chunk local scan (carry-in = 0), emit chunk summaries.
// Thread owns one m (= tid>>2) and 4 consecutive c ((tid&3)*4 + j).
// gamma(m,c) = exp(-|a_m|) * (cos b_c + i sin b_c); reset kills history.
// ---------------------------------------------------------------------------
__global__ __launch_bounds__(256) void k_scanA(
    const float* __restrict__ y, const unsigned char* __restrict__ resets,
    const float* __restrict__ fa, const float* __restrict__ fb,
    float4* __restrict__ summ /* [CH][NPAIR], p = m*16+c */)
{
  __shared__ float gated[CL * MDIM];   // 32 KB
  __shared__ unsigned char rs[CL];
  const int tid = threadIdx.x, chunk = blockIdx.x, t0 = chunk * CL;

  for (int i = tid; i < CL * MDIM; i += 256) {
    const int tl = i >> 6, m = i & 63;
    const size_t t = (size_t)(t0 + tl);
    gated[i] = y[t * NCAT + m] * sigmoidf_(y[t * NCAT + 64 + m]);
  }
  if (tid < CL) rs[tid] = resets[t0 + tid];
  __syncthreads();

  const int m0 = tid >> 2;            // 0..63
  const int cb = (tid & 3) * 4;       // 0,4,8,12
  const float d = __expf(-fabsf(fa[m0]));
  float gre[4], gim[4];
  float sre[4] = {0.f, 0.f, 0.f, 0.f}, sim[4] = {0.f, 0.f, 0.f, 0.f};
  #pragma unroll
  for (int j = 0; j < 4; ++j) {
    gre[j] = d * cosf(fb[cb + j]);
    gim[j] = d * sinf(fb[cb + j]);
  }
  int any = 0;
  for (int tl = 0; tl < CL; ++tl) {
    const int r = rs[tl];
    any |= r;
    const float z = gated[tl * 64 + m0];
    #pragma unroll
    for (int j = 0; j < 4; ++j) {
      const float nre = sre[j] * gre[j] - sim[j] * gim[j] + z;
      const float nim = sre[j] * gim[j] + sim[j] * gre[j];
      sre[j] = r ? z   : nre;
      sim[j] = r ? 0.f : nim;
    }
  }
  #pragma unroll
  for (int j = 0; j < 4; ++j)
    summ[(size_t)chunk * NPAIR + m0 * 16 + cb + j] =
        make_float4(sre[j], sim[j], (float)any, 0.f);
}

// Scan pass B: sequential scan over chunk summaries -> per-chunk carry-in.
__global__ __launch_bounds__(256) void k_scanB(
    const float4* __restrict__ summ, const float* __restrict__ fa,
    const float* __restrict__ fb, float2* __restrict__ carry /* [CH][NPAIR] */)
{
  const int p = blockIdx.x * 256 + threadIdx.x;
  const int m = p >> 4, c = p & 15;
  const float d   = __expf(-fabsf(fa[m]) * (float)CL);
  const float ang = fb[c] * (float)CL;
  const float Gre = d * cosf(ang), Gim = d * sinf(ang);  // gamma^CL
  float cre = 0.f, cim = 0.f;
  for (int j = 0; j < CH; ++j) {
    carry[(size_t)j * NPAIR + p] = make_float2(cre, cim);
    const float4 s = summ[(size_t)j * NPAIR + p];
    if (s.z != 0.f) { cre = s.x; cim = s.y; }     // reset inside chunk
    else {
      const float nre = cre * Gre - cim * Gim + s.x;
      const float nim = cre * Gim + cim * Gre + s.y;
      cre = nre; cim = nim;
    }
  }
}

// Scan pass C: redo local scan seeded with carry, emit z_in as bf16 [T,2048]:
// real -> [t, m*32 + c], imag -> [t, m*32 + 16 + c]. Thread owns (m0, cb..cb+3)
// so each step emits two packed 8-byte stores instead of 8 scalar b16 stores.
// ---------------------------------------------------------------------------
__global__ __launch_bounds__(256) void k_scanC(
    const float* __restrict__ y, const unsigned char* __restrict__ resets,
    const float* __restrict__ fa, const float* __restrict__ fb,
    const float2* __restrict__ carry, __bf16* __restrict__ zin)
{
  __shared__ float gated[CL * MDIM];
  __shared__ unsigned char rs[CL];
  const int tid = threadIdx.x, chunk = blockIdx.x, t0 = chunk * CL;

  for (int i = tid; i < CL * MDIM; i += 256) {
    const int tl = i >> 6, m = i & 63;
    const size_t t = (size_t)(t0 + tl);
    gated[i] = y[t * NCAT + m] * sigmoidf_(y[t * NCAT + 64 + m]);
  }
  if (tid < CL) rs[tid] = resets[t0 + tid];
  __syncthreads();

  const int m0 = tid >> 2;
  const int cb = (tid & 3) * 4;
  const float d = __expf(-fabsf(fa[m0]));
  float gre[4], gim[4], sre[4], sim[4];
  #pragma unroll
  for (int j = 0; j < 4; ++j) {
    gre[j] = d * cosf(fb[cb + j]);
    gim[j] = d * sinf(fb[cb + j]);
    const float2 cv = carry[(size_t)chunk * NPAIR + m0 * 16 + cb + j];
    sre[j] = cv.x; sim[j] = cv.y;
  }
  for (int tl = 0; tl < CL; ++tl) {
    const int r = rs[tl];
    const float z = gated[tl * 64 + m0];
    union { __bf16 h[4]; uint2 u; } pr, pi;
    #pragma unroll
    for (int j = 0; j < 4; ++j) {
      const float nre = sre[j] * gre[j] - sim[j] * gim[j] + z;
      const float nim = sre[j] * gim[j] + sim[j] * gre[j];
      sre[j] = r ? z   : nre;
      sim[j] = r ? 0.f : nim;
      pr.h[j] = (__bf16)sre[j];
      pi.h[j] = (__bf16)sim[j];
    }
    const size_t base = (size_t)(t0 + tl) * KMIX + m0 * 32 + cb;
    *(uint2*)(zin + base)      = pr.u;
    *(uint2*)(zin + base + 16) = pi.u;
  }
}

// ---------------------------------------------------------------------------
// GEMM3: zmix[T,128] = z_in[T,2048](bf16) @ W_mix + b_mix. 64 K-steps, 16
// weight slabs, double-buffered LDS (2 x 32 KB) with async global->LDS;
// 8 N-tiles/wave, 512 WMMAs per wave.
// ---------------------------------------------------------------------------
__global__ __launch_bounds__(256) void k_gemm3(
    const __bf16* __restrict__ zin, const __bf16* __restrict__ Wsw,
    const float* __restrict__ bmix, float* __restrict__ zmix)
{
  __shared__ __bf16 lds[2][4 * 8 * 32 * 16];   // 2 x 32 KB
  const int tid = threadIdx.x, wave = tid >> 5, lane = tid & 31;
  const int rtile = lane & 15, hi = lane >> 4;
  const int rowA = blockIdx.x * 128 + wave * 16 + rtile;
  const __bf16* zr = zin + (size_t)rowA * KMIX;

  v8f acc[8] = {};

#if USE_ASYNC_LDS
  {
    const uint4* src = (const uint4*)Wsw;
    uint4* dst = (uint4*)&lds[0][0];
    #pragma unroll
    for (int i = 0; i < 8; ++i) async_ld16(src + tid + 256 * i, dst + tid + 256 * i);
  }
#endif

  for (int slab = 0; slab < 16; ++slab) {
    const int cur = slab & 1;
#if USE_ASYNC_LDS
    __builtin_amdgcn_s_wait_asynccnt(0);
    __syncthreads();
    if (slab + 1 < 16) {
      const uint4* src = (const uint4*)(Wsw + (size_t)(slab + 1) * (4 * 8 * 32 * 16));
      uint4* dst = (uint4*)&lds[cur ^ 1][0];
      #pragma unroll
      for (int i = 0; i < 8; ++i) async_ld16(src + tid + 256 * i, dst + tid + 256 * i);
    }
#else
    __syncthreads();
    {
      const uint4* src = (const uint4*)(Wsw + (size_t)slab * (4 * 8 * 32 * 16));
      uint4* dst = (uint4*)&lds[cur][0];
      #pragma unroll
      for (int i = 0; i < 8; ++i) dst[tid + 256 * i] = src[tid + 256 * i];
    }
    __syncthreads();
#endif

    #pragma unroll
    for (int ksl = 0; ksl < 4; ++ksl) {
      const int k0 = (slab * 4 + ksl) * 32 + hi * 8;
      BF16Vec a;
      a.q[0] = *(const uint4*)(zr + k0);
      a.q[1] = *(const uint4*)(zr + k0 + 16);
      #pragma unroll
      for (int nt = 0; nt < 8; ++nt) {
        const v16bf B = *(const v16bf*)&lds[cur][((ksl * 8 + nt) * 32 + lane) << 4];
        acc[nt] = __builtin_amdgcn_wmma_f32_16x16x32_bf16(false, a.v, false, B,
                                                          (short)0, acc[nt], false, false);
      }
    }
  }
  const int outRow0 = blockIdx.x * 128 + wave * 16 + hi * 8;
  #pragma unroll
  for (int nt = 0; nt < 8; ++nt) {
    const int col = nt * 16 + rtile;
    const float bias = bmix[col];
    #pragma unroll
    for (int i = 0; i < 8; ++i)
      zmix[(size_t)(outRow0 + i) * ODIM + col] = acc[nt][i] + bias;
  }
}

// ---------------------------------------------------------------------------
// Final: h = zmix * sigmoid(y_gout); LN(h) + y_skip * (1 - gate). One block
// (128 threads) per row.
// ---------------------------------------------------------------------------
__global__ __launch_bounds__(128) void k_final(
    const float* __restrict__ zmix, const float* __restrict__ y,
    float* __restrict__ out)
{
  __shared__ float red[128];
  const int t = blockIdx.x, j = threadIdx.x;
  const float go = sigmoidf_(y[(size_t)t * NCAT + 128 + j]);
  const float sk = y[(size_t)t * NCAT + 256 + j];
  const float h  = zmix[(size_t)t * ODIM + j] * go;

  red[j] = h; __syncthreads();
  for (int s = 64; s > 0; s >>= 1) { if (j < s) red[j] += red[j + s]; __syncthreads(); }
  const float mu = red[0] * (1.0f / 128.0f);
  __syncthreads();
  const float dv = h - mu;
  red[j] = dv * dv; __syncthreads();
  for (int s = 64; s > 0; s >>= 1) { if (j < s) red[j] += red[j + s]; __syncthreads(); }
  const float var = red[0] * (1.0f / 128.0f);

  out[(size_t)t * ODIM + j] = dv * rsqrtf(var + 1e-6f) + sk * (1.0f - go);
}

// ---------------------------------------------------------------------------
extern "C" void kernel_launch(void* const* d_in, const int* in_sizes, int n_in,
                              void* d_out, int out_size, void* d_ws, size_t ws_size,
                              hipStream_t stream) {
  (void)in_sizes; (void)n_in; (void)out_size; (void)ws_size;
  const float* x          = (const float*)d_in[0];
  const unsigned char* rst= (const unsigned char*)d_in[1];
  const float* fa         = (const float*)d_in[2];
  const float* fb         = (const float*)d_in[3];
  const float* W_pre      = (const float*)d_in[4];
  const float* b_pre      = (const float*)d_in[5];
  const float* W_gin      = (const float*)d_in[6];
  const float* b_gin      = (const float*)d_in[7];
  const float* W_gout     = (const float*)d_in[8];
  const float* b_gout     = (const float*)d_in[9];
  const float* W_skip     = (const float*)d_in[10];
  const float* b_skip     = (const float*)d_in[11];
  const float* W_mix      = (const float*)d_in[12];
  const float* b_mix      = (const float*)d_in[13];

  char* ws = (char*)d_ws;
  float*  y     = (float*) (ws);                          // T*384 f32   = 50331648 B
  __bf16* zin   = (__bf16*)(ws + 50331648ull);            // T*2048 bf16 = 134217728 B
  float*  zmix  = (float*) (ws + 184549376ull);           // T*128 f32   = 16777216 B
  __bf16* Wcat  = (__bf16*)(ws + 201326592ull);           // 98304 B
  __bf16* Wmx   = (__bf16*)(ws + 201424896ull);           // 524288 B
  float4* summ  = (float4*)(ws + 201949184ull);           // CH*NPAIR*16 = 4194304 B
  float2* carry = (float2*)(ws + 206143488ull);           // CH*NPAIR*8  = 2097152 B

  k_swizzle_cat<<<192, 256, 0, stream>>>(W_pre, W_gin, W_gout, W_skip, Wcat);
  k_swizzle_mix<<<1024, 256, 0, stream>>>(W_mix, Wmx);
  k_gemm1<<<T_DIM / 128, 256, 0, stream>>>(x, Wcat, b_pre, b_gin, b_gout, b_skip, y);
  k_scanA<<<CH, 256, 0, stream>>>(y, rst, fa, fb, summ);
  k_scanB<<<NPAIR / 256, 256, 0, stream>>>(summ, fa, fb, carry);
  k_scanC<<<CH, 256, 0, stream>>>(y, rst, fa, fb, carry, zin);
  k_gemm3<<<T_DIM / 128, 256, 0, stream>>>(zin, Wmx, b_mix, zmix);
  k_final<<<T_DIM, 128, 0, stream>>>(zmix, y, (float*)d_out);
}
